// BiLSTM_CRF_74612171866591
// MI455X (gfx1250) — matrix operand order repaired and checked by reading, not compiled, not run on hardware
//
#include <hip/hip_runtime.h>

// ---------------- types ----------------
typedef __attribute__((ext_vector_type(16))) __bf16 v16bf;
typedef __attribute__((ext_vector_type(8)))  __bf16 v8bf;
typedef __attribute__((ext_vector_type(8)))  float  v8f;
typedef __attribute__((ext_vector_type(4)))  unsigned int v4u;
typedef __attribute__((ext_vector_type(8)))  int v8i;
typedef __attribute__((ext_vector_type(4)))  int v4i;

union Frag { v16bf v; v8bf h[2]; };

// ---------------- problem constants ----------------
constexpr int Bt  = 128;          // batch
constexpr int Sq  = 512;          // seq len
constexpr int Emb = 300;          // embedding dim
constexpr int EP  = 320;          // padded embedding (mult of 32)
constexpr int Hd  = 256;          // hidden
constexpr int H2  = 512;          // 2*H
constexpr int G4  = 1024;         // 4*H gates
constexpr int Tg  = 9;            // tags
constexpr int Mrows = Bt * Sq;    // 65536

// LDS strides (pad to dodge bank conflicts)
constexpr int HSTR = 264;         // bf16 stride for h
constexpr int GSTR = 1040;        // f32 stride for gate matrix

// LDS layout for lstm_scan (bytes)
constexpr size_t LDS_H   = 0;
constexpr size_t LDS_C   = LDS_H + (size_t)16 * HSTR * 2;
constexpr size_t LDS_G   = LDS_C + (size_t)16 * Hd * 4;
constexpr size_t LDS_XW  = LDS_G + (size_t)16 * GSTR * 4;        // 2 x 32KB xW tiles
constexpr size_t XWBYTES = (size_t)16 * G4 * 2;
constexpr size_t LDS_TOTAL = LDS_XW + 2 * XWBYTES;               // ~153 KB

// ---------------- workspace layout (bytes) ----------------
constexpr size_t OFF_E    = 0;
constexpr size_t SZ_E     = (size_t)Mrows * EP * 2;
constexpr size_t OFF_XWA  = OFF_E + SZ_E;
constexpr size_t SZ_XW    = (size_t)Mrows * G4 * 2;
constexpr size_t OFF_XWB  = OFF_XWA + SZ_XW;
constexpr size_t OFF_L0   = OFF_XWB + SZ_XW;
constexpr size_t SZ_L     = (size_t)Mrows * H2 * 2;
constexpr size_t OFF_L1   = OFF_L0 + SZ_L;
constexpr size_t OFF_EM   = OFF_L1 + SZ_L;
constexpr size_t SZ_EM    = (size_t)Mrows * 16 * 4;
constexpr size_t OFF_W    = OFF_EM + SZ_EM;
constexpr size_t W_IH_L0F = 0;
constexpr size_t W_IH_L0B = W_IH_L0F + (size_t)G4 * EP;
constexpr size_t W_HH_L0F = W_IH_L0B + (size_t)G4 * EP;
constexpr size_t W_HH_L0B = W_HH_L0F + (size_t)G4 * Hd;
constexpr size_t W_IH_L1F = W_HH_L0B + (size_t)G4 * Hd;
constexpr size_t W_IH_L1B = W_IH_L1F + (size_t)G4 * H2;
constexpr size_t W_HH_L1F = W_IH_L1B + (size_t)G4 * H2;
constexpr size_t W_HH_L1B = W_HH_L1F + (size_t)G4 * Hd;
constexpr size_t W_TOTAL  = W_HH_L1B + (size_t)G4 * Hd;
constexpr size_t OFF_LOSS = OFF_W + W_TOTAL * 2;

#if __has_builtin(__builtin_amdgcn_tensor_load_to_lds) && __has_builtin(__builtin_amdgcn_s_wait_tensorcnt)
#define USE_TDM 1
#endif

__device__ __forceinline__ float sigf(float x) { return 1.0f / (1.0f + __expf(-x)); }

#if defined(USE_TDM)
// TDM: DMA a 16-row x 1024-col bf16 tile (row stride Sq*G4 elements) into LDS.
__device__ __forceinline__ void tdm_load_xw_tile(const __bf16* gaddr, unsigned lds_addr) {
    unsigned long long ga = (unsigned long long)(uintptr_t)gaddr;
    v4u g0;
    g0[0] = 1u;                                             // count=1, user mode
    g0[1] = lds_addr;                                       // LDS byte address
    g0[2] = (unsigned)(ga & 0xFFFFFFFFu);                   // global_addr[31:0]
    g0[3] = (unsigned)((ga >> 32) & 0x01FFFFFFu) | (2u << 30); // addr[56:32] | type=2
    v8i g1;
    g1[0] = 0x10000;                                        // data_size=1 (2 bytes)
    g1[1] = (int)(((unsigned)G4 & 0xFFFFu) << 16);          // tensor_dim0[15:0]=1024
    g1[2] = (int)(16u << 16);                               // tensor_dim1[15:0]=16
    g1[3] = (int)(((unsigned)G4 & 0xFFFFu) << 16);          // tile_dim0=1024
    g1[4] = 16;                                             // tile_dim1=16, tile_dim2=0
    g1[5] = (int)((unsigned)(Sq * G4));                     // tensor_dim0_stride lo
    g1[6] = 0;
    g1[7] = 0;
    v4i z4 = {0, 0, 0, 0};
#if __clang_major__ >= 23
    v8i z8 = {0, 0, 0, 0, 0, 0, 0, 0};
    __builtin_amdgcn_tensor_load_to_lds(g0, g1, z4, z4, z8, 0);
#else
    __builtin_amdgcn_tensor_load_to_lds(g0, g1, z4, z4, 0);
#endif
}
#endif

// ---------------- weight convert f32 -> bf16 with K padding ----------------
__global__ void convert_pad(const float* __restrict__ src, __bf16* __restrict__ dst,
                            int N, int K, int Kp) {
    int idx = blockIdx.x * 256 + threadIdx.x;
    int total = N * Kp;
    if (idx >= total) return;
    int n = idx / Kp, k = idx - n * Kp;
    float v = (k < K) ? src[(size_t)n * K + k] : 0.0f;
    dst[idx] = (__bf16)v;
}

// ---------------- embedding gather + cvt to bf16 ----------------
__global__ void embed_gather(const int* __restrict__ x, const float* __restrict__ emb,
                             __bf16* __restrict__ e) {
    int idx = blockIdx.x * 256 + threadIdx.x;
    const int total = Mrows * EP;
    if (idx >= total) return;
    int m = idx / EP, k = idx - m * EP;
    float v = 0.0f;
    if (k < Emb) {
        int tok = x[m];
        v = emb[(size_t)tok * Emb + k];
    }
    e[idx] = (__bf16)v;
}

// ---------------- bf16 WMMA GEMM: C[M,N] = A[M,K] * W[N,K]^T + bias ----------------
// K is compile-time: full unroll, immediate-folded addresses, no pointer spills.
// one wave per block; 16(M) x 64(N) tile; 2-deep software pipeline over K (K%64==0)
template <int K>
__global__ void gemm_bf16(const __bf16* __restrict__ A, const __bf16* __restrict__ W,
                          const float* __restrict__ bias, __bf16* __restrict__ Cout,
                          int N) {
    const int lane = threadIdx.x;
    const int m0   = blockIdx.y << 4;
    const int n0b  = blockIdx.x << 6;          // 64 cols per block
    const int klo  = (lane >> 4) << 3;         // A frag: 0 or 8
    const int kbs  = (lane >> 4) << 3;         // B frag: 0 or 8
    const int nc   = lane & 15;

    const __bf16* Arow  = A + (size_t)(m0 + (lane & 15)) * K + klo;
    const __bf16* Wbase = W + (size_t)(n0b + nc) * K + kbs;   // n-tile offsets fold to immediates

    v8f acc[4] = {};
    auto loadA = [&](Frag& a, int k0) {
        a.h[0] = *(const v8bf*)(Arow + k0);
        a.h[1] = *(const v8bf*)(Arow + k0 + 16);
    };
    auto loadB = [&](Frag* b, int k0) {
#pragma unroll
        for (int nt = 0; nt < 4; ++nt) {
            b[nt].h[0] = *(const v8bf*)(Wbase + (size_t)nt * 16 * K + k0);
            b[nt].h[1] = *(const v8bf*)(Wbase + (size_t)nt * 16 * K + k0 + 16);
        }
    };
    auto mma = [&](Frag& a, Frag* b) {
#pragma unroll
        for (int nt = 0; nt < 4; ++nt)
            acc[nt] = __builtin_amdgcn_wmma_f32_16x16x32_bf16(
                false, a.v, false, b[nt].v, (short)0, acc[nt], false, false);
    };

    Frag a0, a1, b0[4], b1[4];
    loadA(a0, 0); loadB(b0, 0);
#pragma unroll
    for (int k0 = 0; k0 < K; k0 += 64) {
        loadA(a1, k0 + 32); loadB(b1, k0 + 32);
        if (k0 + 128 <= K) __builtin_prefetch(Arow + k0 + 128, 0, 3);
        mma(a0, b0);
        if (k0 + 64 < K) { loadA(a0, k0 + 64); loadB(b0, k0 + 64); }
        mma(a1, b1);
    }

    const int mh = (lane >> 4) << 3;
#pragma unroll
    for (int nt = 0; nt < 4; ++nt) {
        int n = n0b + (nt << 4) + nc;
        float bv = bias[n];
#pragma unroll
        for (int r = 0; r < 8; ++r)
            Cout[(size_t)(m0 + mh + r) * N + n] = (__bf16)(acc[nt][r] + bv);
    }
}

// ---------------- persistent bidirectional LSTM scan ----------------
// grid (8 batch tiles, 2 dirs), 256 threads (8 waves). State h/c live in LDS.
// Per step: TDM prefetches xW[t+1] tile into LDS while WMMA computes h@Whh^T.
// Wave covers 128 N cols, processed as two halves of 4 n-tiles (register fit).
// Whh streams from L2 every step (shared by all WGs). An opaque zero-offset
// (asm) blocks LICM from hoisting the loop-invariant weight loads while
// keeping pointer provenance so loads lower to global_load (not flat_load).
__global__ void lstm_scan(const __bf16* __restrict__ xW_f, const __bf16* __restrict__ xW_b,
                          const __bf16* __restrict__ Whh_f, const __bf16* __restrict__ Whh_b,
                          __bf16* __restrict__ out /* [Mrows][H2] */) {
    extern __shared__ char smem[];
    __bf16* h_sm  = (__bf16*)(smem + LDS_H);
    float*  c_sm  = (float*)(smem + LDS_C);
    float*  g_sm  = (float*)(smem + LDS_G);
    __bf16* xw_sm = (__bf16*)(smem + LDS_XW);

    const int dir = blockIdx.y;
    const __bf16* xW  = dir ? xW_b  : xW_f;
    const __bf16* Whh = dir ? Whh_b : Whh_f;
    const int b0  = blockIdx.x << 4;
    const int tid = threadIdx.x;
    const int lane = tid & 31;
    const int w    = tid >> 5;

    for (int i = tid; i < 16 * HSTR; i += 256) h_sm[i] = (__bf16)0.0f;
    for (int i = tid; i < 16 * Hd;   i += 256) c_sm[i] = 0.0f;
    __syncthreads();

    const int arow = lane & 15;
    const int klo  = (lane >> 4) << 3;
    const int kbs  = (lane >> 4) << 3;
    const int nc   = lane & 15;
    const int mh   = (lane >> 4) << 3;

#if defined(USE_TDM)
    const unsigned xw_lds0 = (unsigned)(uintptr_t)(void*)xw_sm;
    // prime the pipeline: tile for t=0
    if (w == 0) {
        int t0 = dir ? (Sq - 1) : 0;
        tdm_load_xw_tile(xW + ((size_t)b0 * Sq + t0) * G4, xw_lds0);
    }
#endif

    for (int t = 0; t < Sq; ++t) {
        const int ta = dir ? (Sq - 1 - t) : t;

#if defined(USE_TDM)
        if (w == 0 && t + 1 < Sq) {
            int tn = dir ? (Sq - 2 - t) : (t + 1);
            tdm_load_xw_tile(xW + ((size_t)b0 * Sq + tn) * G4,
                             xw_lds0 + (unsigned)(((t + 1) & 1) * XWBYTES));
        }
#endif

        // ---- A frags (h state) from LDS, reused across both N halves ----
        Frag af[8];
#pragma unroll
        for (int kk = 0; kk < 8; ++kk) {
            const __bf16* hp = h_sm + arow * HSTR + kk * 32 + klo;
            af[kk].h[0] = *(const v8bf*)(hp);
            af[kk].h[1] = *(const v8bf*)(hp + 16);
        }

        // ---- h(16x256) @ Whh^T(256x1024): two 4-n-tile halves, pipelined B ----
#pragma unroll
        for (int half = 0; half < 2; ++half) {
            const int nbase = (w << 7) + (half << 6);
            // opaque zero offset: defeats LICM (loads stay in the t-loop) but
            // preserves Whh provenance so codegen picks global_load_b128.
            int koff = 0;
            asm volatile("" : "+v"(koff));
            const __bf16* WB = Whh + (size_t)(nbase + nc) * Hd + kbs + koff;

            v8f acc[4] = {};
            auto loadB = [&](Frag* b, int kk) {
#pragma unroll
                for (int nt = 0; nt < 4; ++nt) {
                    b[nt].h[0] = *(const v8bf*)(WB + (size_t)nt * 16 * Hd + kk * 32);
                    b[nt].h[1] = *(const v8bf*)(WB + (size_t)nt * 16 * Hd + kk * 32 + 16);
                }
            };
            auto mmaK = [&](Frag& a, Frag* b) {
#pragma unroll
                for (int nt = 0; nt < 4; ++nt)
                    acc[nt] = __builtin_amdgcn_wmma_f32_16x16x32_bf16(
                        false, a.v, false, b[nt].v, (short)0, acc[nt], false, false);
            };
            Frag b0f[4], b1f[4];
            loadB(b0f, 0);
#pragma unroll
            for (int kk = 0; kk < 8; kk += 2) {
                loadB(b1f, kk + 1);
                mmaK(af[kk], b0f);
                if (kk + 2 < 8) loadB(b0f, kk + 2);
                mmaK(af[kk + 1], b1f);
            }
#pragma unroll
            for (int nt = 0; nt < 4; ++nt) {
                int n = nbase + (nt << 4) + nc;
#pragma unroll
                for (int r = 0; r < 8; ++r) g_sm[(mh + r) * GSTR + n] = acc[nt][r];
            }
        }

#if defined(USE_TDM)
        if (w == 0) {
            if (t + 1 < Sq) __builtin_amdgcn_s_wait_tensorcnt(1);
            else            __builtin_amdgcn_s_wait_tensorcnt(0);
        }
#endif
        __syncthreads();

        // ---- pointwise LSTM cell: thread tid owns hidden unit j for 16 rows ----
        const int j = tid;
#if defined(USE_TDM)
        const __bf16* xt = xw_sm + (size_t)(t & 1) * 16 * G4;
#endif
#pragma unroll 4
        for (int m = 0; m < 16; ++m) {
            size_t rg = (size_t)(b0 + m) * Sq + ta;
#if defined(USE_TDM)
            const __bf16* xr = xt + m * G4;
#else
            const __bf16* xr = xW + rg * G4;
#endif
            float gi = g_sm[m * GSTR +       j] + (float)xr[      j];
            float gf = g_sm[m * GSTR + 256 + j] + (float)xr[256 + j];
            float gg = g_sm[m * GSTR + 512 + j] + (float)xr[512 + j];
            float go = g_sm[m * GSTR + 768 + j] + (float)xr[768 + j];
            float c  = sigf(gf) * c_sm[m * Hd + j] + sigf(gi) * tanhf(gg);
            float hh = sigf(go) * tanhf(c);
            c_sm[m * Hd + j] = c;
            h_sm[m * HSTR + j] = (__bf16)hh;
            out[rg * H2 + (dir << 8) + j] = (__bf16)hh;
        }
        __syncthreads();
    }
}

// ---------------- FC: emissions ----------------
__global__ void fc_emissions(const __bf16* __restrict__ l1, const float* __restrict__ wt,
                             const float* __restrict__ bs, float* __restrict__ em) {
    int gid = blockIdx.x * 256 + threadIdx.x;
    int tt = gid & 15, m = gid >> 4;
    if (m >= Mrows) return;
    float s = 0.0f;
    if (tt < Tg) {
        const __bf16* xr = l1 + (size_t)m * H2;
        const float*  wr = wt + tt * H2;
#pragma unroll 8
        for (int k = 0; k < H2; ++k) s += (float)xr[k] * wr[k];
        s += bs[tt];
    }
    em[(size_t)m * 16 + tt] = s;
}

// ---------------- CRF NLL: one wave per batch element ----------------
__global__ void crf_nll(const float* __restrict__ em, const int* __restrict__ x,
                        const int* __restrict__ tags, const float* __restrict__ start,
                        const float* __restrict__ endv, const float* __restrict__ trans,
                        float* __restrict__ loss) {
    const int b = blockIdx.x;
    const int lane = threadIdx.x;
    const bool valid = lane < Tg;
    const float* emB = em + (size_t)b * Sq * 16;

    float trcol[Tg];
#pragma unroll
    for (int i = 0; i < Tg; ++i) trcol[i] = valid ? trans[i * Tg + lane] : -1e30f;

    float alpha = valid ? (start[lane] + emB[lane]) : -1e30f;

    float num = 0.0f;
    int prev = 0, cnt = (x[b * Sq] != 0) ? 1 : 0;
    if (lane == 0) {
        prev = tags[b * Sq];
        num = start[prev] + emB[prev];
    }

    for (int t = 1; t < Sq; ++t) {
        bool mt = (x[b * Sq + t] != 0);
        float v[Tg], mx = -1e30f;
#pragma unroll
        for (int i = 0; i < Tg; ++i) {
            float ai = __shfl(alpha, i, 32);
            v[i] = ai + trcol[i];
            mx = fmaxf(mx, v[i]);
        }
        float s = 0.0f;
#pragma unroll
        for (int i = 0; i < Tg; ++i) s += __expf(v[i] - mx);
        float nxt = mx + __logf(s) + (valid ? emB[t * 16 + lane] : 0.0f);
        if (mt && valid) alpha = nxt;
        if (lane == 0) {
            int cur = tags[b * Sq + t];
            if (mt) { num += trans[prev * Tg + cur] + emB[t * 16 + cur]; cnt++; }
            prev = cur;
        }
    }

    float vv = valid ? (alpha + endv[lane]) : -1e30f;
    float mx = vv;
#pragma unroll
    for (int off = 16; off >= 1; off >>= 1) mx = fmaxf(mx, __shfl_xor(mx, off, 32));
    float s = __expf(vv - mx);
#pragma unroll
    for (int off = 16; off >= 1; off >>= 1) s += __shfl_xor(s, off, 32);
    float denom = mx + __logf(s);

    if (lane == 0) {
        int last = cnt - 1; if (last < 0) last = 0;
        num += endv[tags[b * Sq + last]];
        loss[b] = denom - num;
    }
}

__global__ void reduce_mean(const float* __restrict__ loss, float* __restrict__ out) {
    __shared__ float sm[Bt];
    int t = threadIdx.x;
    sm[t] = loss[t];
    __syncthreads();
    if (t == 0) {
        float s = 0.0f;
        for (int i = 0; i < Bt; ++i) s += sm[i];
        out[0] = s / (float)Bt;
    }
}

// ---------------- host-side launch ----------------
extern "C" void kernel_launch(void* const* d_in, const int* in_sizes, int n_in,
                              void* d_out, int out_size, void* d_ws, size_t ws_size,
                              hipStream_t stream) {
    (void)in_sizes; (void)n_in; (void)out_size; (void)ws_size;
    const int*   x        = (const int*)d_in[0];
    const int*   tags     = (const int*)d_in[1];
    const float* emb      = (const float*)d_in[2];
    const float* Wih_l0f  = (const float*)d_in[3];
    const float* Whh_l0f  = (const float*)d_in[4];
    const float* b_l0f    = (const float*)d_in[5];
    const float* Wih_l0b  = (const float*)d_in[6];
    const float* Whh_l0b  = (const float*)d_in[7];
    const float* b_l0b    = (const float*)d_in[8];
    const float* Wih_l1f  = (const float*)d_in[9];
    const float* Whh_l1f  = (const float*)d_in[10];
    const float* b_l1f    = (const float*)d_in[11];
    const float* Wih_l1b  = (const float*)d_in[12];
    const float* Whh_l1b  = (const float*)d_in[13];
    const float* b_l1b    = (const float*)d_in[14];
    const float* fc_w     = (const float*)d_in[15];
    const float* fc_b     = (const float*)d_in[16];
    const float* crf_s    = (const float*)d_in[17];
    const float* crf_e    = (const float*)d_in[18];
    const float* crf_t    = (const float*)d_in[19];

    char* ws = (char*)d_ws;
    __bf16* eB   = (__bf16*)(ws + OFF_E);
    __bf16* xwA  = (__bf16*)(ws + OFF_XWA);
    __bf16* xwB  = (__bf16*)(ws + OFF_XWB);
    __bf16* l0   = (__bf16*)(ws + OFF_L0);
    __bf16* l1   = (__bf16*)(ws + OFF_L1);
    float*  emis = (float*)(ws + OFF_EM);
    __bf16* wp   = (__bf16*)(ws + OFF_W);
    float*  loss = (float*)(ws + OFF_LOSS);

    __bf16* w_ih_l0f = wp + W_IH_L0F;
    __bf16* w_ih_l0b = wp + W_IH_L0B;
    __bf16* w_hh_l0f = wp + W_HH_L0F;
    __bf16* w_hh_l0b = wp + W_HH_L0B;
    __bf16* w_ih_l1f = wp + W_IH_L1F;
    __bf16* w_ih_l1b = wp + W_IH_L1B;
    __bf16* w_hh_l1f = wp + W_HH_L1F;
    __bf16* w_hh_l1b = wp + W_HH_L1B;

    auto cdiv = [](int a, int b) { return (a + b - 1) / b; };

    convert_pad<<<cdiv(G4 * EP, 256), 256, 0, stream>>>(Wih_l0f, w_ih_l0f, G4, Emb, EP);
    convert_pad<<<cdiv(G4 * EP, 256), 256, 0, stream>>>(Wih_l0b, w_ih_l0b, G4, Emb, EP);
    convert_pad<<<cdiv(G4 * Hd, 256), 256, 0, stream>>>(Whh_l0f, w_hh_l0f, G4, Hd, Hd);
    convert_pad<<<cdiv(G4 * Hd, 256), 256, 0, stream>>>(Whh_l0b, w_hh_l0b, G4, Hd, Hd);
    convert_pad<<<cdiv(G4 * H2, 256), 256, 0, stream>>>(Wih_l1f, w_ih_l1f, G4, H2, H2);
    convert_pad<<<cdiv(G4 * H2, 256), 256, 0, stream>>>(Wih_l1b, w_ih_l1b, G4, H2, H2);
    convert_pad<<<cdiv(G4 * Hd, 256), 256, 0, stream>>>(Whh_l1f, w_hh_l1f, G4, Hd, Hd);
    convert_pad<<<cdiv(G4 * Hd, 256), 256, 0, stream>>>(Whh_l1b, w_hh_l1b, G4, Hd, Hd);

    embed_gather<<<cdiv(Mrows * EP, 256), 256, 0, stream>>>(x, emb, eB);

    const dim3 gGrid(G4 / 64, Mrows / 16);

    gemm_bf16<EP><<<gGrid, 32, 0, stream>>>(eB, w_ih_l0f, b_l0f, xwA, G4);
    gemm_bf16<EP><<<gGrid, 32, 0, stream>>>(eB, w_ih_l0b, b_l0b, xwB, G4);

    lstm_scan<<<dim3(Bt / 16, 2), 256, LDS_TOTAL, stream>>>(xwA, xwB, w_hh_l0f, w_hh_l0b, l0);

    gemm_bf16<H2><<<gGrid, 32, 0, stream>>>(l0, w_ih_l1f, b_l1f, xwA, G4);
    gemm_bf16<H2><<<gGrid, 32, 0, stream>>>(l0, w_ih_l1b, b_l1b, xwB, G4);

    lstm_scan<<<dim3(Bt / 16, 2), 256, LDS_TOTAL, stream>>>(xwA, xwB, w_hh_l1f, w_hh_l1b, l1);

    fc_emissions<<<cdiv(Mrows * 16, 256), 256, 0, stream>>>(l1, fc_w, fc_b, emis);

    crf_nll<<<Bt, 32, 0, stream>>>(emis, x, tags, crf_s, crf_e, crf_t, loss);

    reduce_mean<<<1, Bt, 0, stream>>>(loss, (float*)d_out);
}
